// GeneGAT_2680059593394
// MI455X (gfx1250) — matrix operand order repaired and checked by reading, not compiled, not run on hardware
//
#include <hip/hip_runtime.h>
#include <hip/hip_bf16.h>

#define N_NODES 50000
#define E_EDGES 800000
#define ET      (E_EDGES + N_NODES)   // 850000 edges incl. self loops
#define DIN     256
#define HCDIM   256                   // 4 heads * 64
#define NCOLS   768                   // xl | xr | skip fused GEMM width
#define NTILES  48                    // 768/16
#define NGROUPS 6                     // 48/8 : each wave owns 8 n-tiles
#define KSTEPS  8                     // 256/32
#define NEG_SLOPE 0.2f
#define LN_EPS  1e-5f

typedef __bf16 bf16;
typedef __attribute__((ext_vector_type(16))) __bf16 v16bf;
typedef __attribute__((ext_vector_type(8)))  __bf16 v8bf;
typedef __attribute__((ext_vector_type(8)))  float  v8f;

// ---- ordered-int encoding for float atomic max -----------------------------
__device__ __forceinline__ int ford(float f) {
    int i = __float_as_int(f);
    return i >= 0 ? i : (i ^ 0x7fffffff);
}
__device__ __forceinline__ float ordf(int k) {
    return __int_as_float(k >= 0 ? k : (k ^ 0x7fffffff));
}

// ---- 0) init accumulators / seed d_out with bias2 --------------------------
__global__ __launch_bounds__(256)
void k_init(float* h1acc, int* m1, float* den1, int* m2, float* den2,
            float* out, const float* bias2) {
    int t = blockIdx.x * 256 + threadIdx.x;              // grid covers N*256
    if (t < N_NODES * HCDIM) h1acc[t] = 0.f;
    if (t < N_NODES * 4) { m1[t] = (int)0x80000000; den1[t] = 0.f; }
    if (t < N_NODES)     { m2[t] = (int)0x80000000; den2[t] = 0.f; }
    if (t < N_NODES * 2) out[t] = bias2[t & 1];
}

// ---- 1) x (f32) -> bf16 ----------------------------------------------------
__global__ __launch_bounds__(256)
void k_cvt_x(const float* __restrict__ x, bf16* __restrict__ xbf) {
    int t = blockIdx.x * 256 + threadIdx.x;
    if (t < N_NODES * DIN) xbf[t] = (bf16)x[t];
}

// ---- 2) pack fused weight [256 x 768] into WMMA B-fragment layout ----------
// dest idx = ((s*48 + j)*32 + lane)*16 + i  ->  Wcat[k][col],
// k = s*32 + (lane>>4)*8 + (i&7) + (i>>3)*16, col = j*16 + (lane&15)
__global__ __launch_bounds__(256)
void k_pack_w(const float* __restrict__ W1l, const float* __restrict__ W1r,
              const float* __restrict__ Wsk, bf16* __restrict__ wp) {
    int t = blockIdx.x * 256 + threadIdx.x;
    if (t >= KSTEPS * NTILES * 32 * 16) return;
    int i = t & 15, u = t >> 4;
    int L = u & 31;  u >>= 5;
    int j = u % NTILES;
    int s = u / NTILES;
    int k   = s * 32 + ((L >> 4) << 3) + (i & 7) + ((i >> 3) << 4);
    int col = j * 16 + (L & 15);
    float v;
    if (col < 256)      v = W1l[k * 256 + col];
    else if (col < 512) v = W1r[k * 256 + (col - 256)];
    else                v = Wsk[k * 256 + (col - 512)];
    wp[t] = (bf16)v;
}

// ---- 3) fused GEMM: [50000x256]bf16 x [256x768]bf16 -> f32 (+bias) ---------
// one wave per (16-row tile x 8 n-tiles): A fragment loaded once per k-step
// and reused across 8 v_wmma_f32_16x16x32_bf16 -> 64 WMMA per wave.
__global__ __launch_bounds__(256)
void k_gemm1(const bf16* __restrict__ xbf, const bf16* __restrict__ wp,
             const float* __restrict__ b1l, const float* __restrict__ b1r,
             const float* __restrict__ bsk, float* __restrict__ XLRS) {
    int gw = blockIdx.x * 8 + (threadIdx.x >> 5);        // global wave id
    if (gw >= (N_NODES / 16) * NGROUPS) return;          // 18750 waves
    int mtile = gw / NGROUPS;
    int nbase = (gw - mtile * NGROUPS) * 8;              // first of 8 n-tiles
    int lane = threadIdx.x & 31;
    int m16 = lane & 15, khalf = lane >> 4;
    const bf16* xrow = xbf + (size_t)(mtile * 16 + m16) * DIN;
    v8f acc[8] = {};
#pragma unroll
    for (int s = 0; s < KSTEPS; ++s) {
        int k0 = s * 32 + khalf * 8;
        v8bf alo = *(const v8bf*)(xrow + k0);
        v8bf ahi = *(const v8bf*)(xrow + k0 + 16);
        v16bf a;
#pragma unroll
        for (int i = 0; i < 8; ++i) { a[i] = alo[i]; a[i + 8] = ahi[i]; }
        const bf16* bp = wp + (((size_t)(s * NTILES + nbase) * 32 + lane) << 4);
#pragma unroll
        for (int j = 0; j < 8; ++j) {
            v16bf b = *(const v16bf*)(bp + ((size_t)j * 32 << 4));
            acc[j] = __builtin_amdgcn_wmma_f32_16x16x32_bf16(
                         false, a, false, b, (short)0, acc[j], false, false);
        }
    }
    int rowbase = mtile * 16 + khalf * 8;
#pragma unroll
    for (int j = 0; j < 8; ++j) {
        int col = (nbase + j) * 16 + m16;
        float bv = col < 256 ? b1l[col]
                             : (col < 512 ? b1r[col - 256] : bsk[col - 512]);
#pragma unroll
        for (int r = 0; r < 8; ++r)
            XLRS[(size_t)(rowbase + r) * NCOLS + col] = acc[j][r] + bv;
    }
}

// ---- 4) conv1 edge logits + segment max (wave per edge) --------------------
__global__ __launch_bounds__(256)
void k_edge1_logits(const int* __restrict__ ei, const float* __restrict__ XLRS,
                    const float* __restrict__ att1, float* __restrict__ lg1,
                    int* __restrict__ m1) {
    int e = blockIdx.x * 8 + (threadIdx.x >> 5);
    if (e >= ET) return;
    int lane = threadIdx.x & 31;
    int s, d;
    if (e < E_EDGES) { s = ei[e]; d = ei[E_EDGES + e]; } else { s = d = e - E_EDGES; }
    const float* xl = XLRS + (size_t)s * NCOLS;
    const float* xr = XLRS + (size_t)d * NCOLS + 256;
    int c0 = lane * 8;
    __builtin_prefetch(xl + c0, 0, 1);            // global_prefetch_b8
    __builtin_prefetch(xr + c0, 0, 1);
    float4 l0 = *(const float4*)(xl + c0), l1 = *(const float4*)(xl + c0 + 4);
    float4 r0 = *(const float4*)(xr + c0), r1 = *(const float4*)(xr + c0 + 4);
    float lv[8] = {l0.x, l0.y, l0.z, l0.w, l1.x, l1.y, l1.z, l1.w};
    float rv[8] = {r0.x, r0.y, r0.z, r0.w, r1.x, r1.y, r1.z, r1.w};
    float sum = 0.f;
#pragma unroll
    for (int i = 0; i < 8; ++i) {
        float v = lv[i] + rv[i];
        v = v > 0.f ? v : NEG_SLOPE * v;
        sum += v * att1[c0 + i];
    }
    sum += __shfl_xor(sum, 1); sum += __shfl_xor(sum, 2); sum += __shfl_xor(sum, 4);
    if ((lane & 7) == 0) {                      // lanes 0,8,16,24 -> heads 0..3
        int h = lane >> 3;
        lg1[(size_t)e * 4 + h] = sum;
        atomicMax(m1 + d * 4 + h, ford(sum));
    }
}

// ---- 5) conv1 softmax denominators (thread per edge*head) ------------------
__global__ __launch_bounds__(256)
void k_edge1_denom(const int* __restrict__ ei, float* __restrict__ lg1,
                   const int* __restrict__ m1, float* __restrict__ den1) {
    int t = blockIdx.x * 256 + threadIdx.x;
    if (t >= ET * 4) return;
    int e = t >> 2, h = t & 3;
    int d = (e < E_EDGES) ? ei[E_EDGES + e] : (e - E_EDGES);
    float a = __expf(lg1[t] - ordf(m1[d * 4 + h]));
    lg1[t] = a;                                  // reuse buffer: store numerator
    atomicAdd(den1 + d * 4 + h, a);
}

// ---- 6) conv1 weighted aggregation (wave per edge) -------------------------
__global__ __launch_bounds__(256)
void k_edge1_agg(const int* __restrict__ ei, const float* __restrict__ XLRS,
                 const float* __restrict__ lg1, const float* __restrict__ den1,
                 float* __restrict__ h1acc) {
    int e = blockIdx.x * 8 + (threadIdx.x >> 5);
    if (e >= ET) return;
    int lane = threadIdx.x & 31;
    int s, d;
    if (e < E_EDGES) { s = ei[e]; d = ei[E_EDGES + e]; } else { s = d = e - E_EDGES; }
    int h = lane >> 3, c0 = lane * 8;
    const float* xl = XLRS + (size_t)s * NCOLS;
    __builtin_prefetch(xl + c0, 0, 1);            // global_prefetch_b8
    float alpha = lg1[(size_t)e * 4 + h] / den1[d * 4 + h];
    float4 l0 = *(const float4*)(xl + c0), l1 = *(const float4*)(xl + c0 + 4);
    float lv[8] = {l0.x, l0.y, l0.z, l0.w, l1.x, l1.y, l1.z, l1.w};
    float* dstp = h1acc + (size_t)d * HCDIM + c0;
#pragma unroll
    for (int i = 0; i < 8; ++i) atomicAdd(dstp + i, lv[i] * alpha);
}

// ---- 7) h = elu(LN(conv1 + bias1 + skip)) ; one block per node -------------
__global__ __launch_bounds__(256)
void k_node_fuse(float* __restrict__ h1acc, const float* __restrict__ XLRS,
                 const float* __restrict__ bias1, const float* __restrict__ gamma,
                 const float* __restrict__ beta) {
    __shared__ float red[8];
    int n = blockIdx.x, c = threadIdx.x;
    float v = h1acc[(size_t)n * HCDIM + c] + bias1[c] + XLRS[(size_t)n * NCOLS + 512 + c];
    // mean
    float t = v;
    for (int o = 16; o; o >>= 1) t += __shfl_xor(t, o);
    if ((c & 31) == 0) red[c >> 5] = t;
    __syncthreads();
    if (c == 0) { float a = 0; for (int i = 0; i < 8; ++i) a += red[i]; red[0] = a; }
    __syncthreads();
    float mu = red[0] * (1.f / 256.f);
    __syncthreads();
    // variance
    float dvi = v - mu;
    t = dvi * dvi;
    for (int o = 16; o; o >>= 1) t += __shfl_xor(t, o);
    if ((c & 31) == 0) red[c >> 5] = t;
    __syncthreads();
    if (c == 0) { float a = 0; for (int i = 0; i < 8; ++i) a += red[i]; red[0] = a; }
    __syncthreads();
    float var = red[0] * (1.f / 256.f);
    float y = dvi * rsqrtf(var + LN_EPS) * gamma[c] + beta[c];
    y = y > 0.f ? y : (__expf(y) - 1.f);         // ELU
    h1acc[(size_t)n * HCDIM + c] = y;            // reuse as h
}

// ---- 8) conv2 linear transforms (wave per node): h[256] -> xl2/xr2 [2] -----
__global__ __launch_bounds__(256)
void k_conv2_lin(const float* __restrict__ h, const float* __restrict__ W2l,
                 const float* __restrict__ b2l, const float* __restrict__ W2r,
                 const float* __restrict__ b2r, float* __restrict__ xl2,
                 float* __restrict__ xr2) {
    int n = blockIdx.x * 8 + (threadIdx.x >> 5);
    if (n >= N_NODES) return;
    int lane = threadIdx.x & 31, c0 = lane * 8;
    const float* hp = h + (size_t)n * HCDIM;
    float s00 = 0, s01 = 0, s10 = 0, s11 = 0;
#pragma unroll
    for (int i = 0; i < 8; ++i) {
        float hv = hp[c0 + i];
        int c = c0 + i;
        s00 += hv * W2l[c * 2 + 0]; s01 += hv * W2l[c * 2 + 1];
        s10 += hv * W2r[c * 2 + 0]; s11 += hv * W2r[c * 2 + 1];
    }
    for (int o = 16; o; o >>= 1) {
        s00 += __shfl_xor(s00, o); s01 += __shfl_xor(s01, o);
        s10 += __shfl_xor(s10, o); s11 += __shfl_xor(s11, o);
    }
    if (lane == 0) {
        xl2[n * 2 + 0] = s00 + b2l[0]; xl2[n * 2 + 1] = s01 + b2l[1];
        xr2[n * 2 + 0] = s10 + b2r[0]; xr2[n * 2 + 1] = s11 + b2r[1];
    }
}

// ---- 9) conv2 edge logits + segment max (thread per edge) ------------------
__global__ __launch_bounds__(256)
void k_edge2_logits(const int* __restrict__ ei, const float* __restrict__ xl2,
                    const float* __restrict__ xr2, const float* __restrict__ att2,
                    float* __restrict__ lg2, int* __restrict__ m2) {
    int e = blockIdx.x * 256 + threadIdx.x;
    if (e >= ET) return;
    int s, d;
    if (e < E_EDGES) { s = ei[e]; d = ei[E_EDGES + e]; } else { s = d = e - E_EDGES; }
    float v0 = xl2[s * 2 + 0] + xr2[d * 2 + 0];
    float v1 = xl2[s * 2 + 1] + xr2[d * 2 + 1];
    v0 = v0 > 0.f ? v0 : NEG_SLOPE * v0;
    v1 = v1 > 0.f ? v1 : NEG_SLOPE * v1;
    float lg = v0 * att2[0] + v1 * att2[1];
    lg2[e] = lg;
    atomicMax(m2 + d, ford(lg));
}

// ---- 10) conv2 denominators ------------------------------------------------
__global__ __launch_bounds__(256)
void k_edge2_denom(const int* __restrict__ ei, float* __restrict__ lg2,
                   const int* __restrict__ m2, float* __restrict__ den2) {
    int e = blockIdx.x * 256 + threadIdx.x;
    if (e >= ET) return;
    int d = (e < E_EDGES) ? ei[E_EDGES + e] : (e - E_EDGES);
    float a = __expf(lg2[e] - ordf(m2[d]));
    lg2[e] = a;
    atomicAdd(den2 + d, a);
}

// ---- 11) conv2 aggregation straight into d_out (seeded with bias2) ---------
__global__ __launch_bounds__(256)
void k_edge2_agg(const int* __restrict__ ei, const float* __restrict__ xl2,
                 const float* __restrict__ lg2, const float* __restrict__ den2,
                 float* __restrict__ out) {
    int e = blockIdx.x * 256 + threadIdx.x;
    if (e >= ET) return;
    int s, d;
    if (e < E_EDGES) { s = ei[e]; d = ei[E_EDGES + e]; } else { s = d = e - E_EDGES; }
    float alpha = lg2[e] / den2[d];
    atomicAdd(out + d * 2 + 0, xl2[s * 2 + 0] * alpha);
    atomicAdd(out + d * 2 + 1, xl2[s * 2 + 1] * alpha);
}

extern "C" void kernel_launch(void* const* d_in, const int* in_sizes, int n_in,
                              void* d_out, int out_size, void* d_ws, size_t ws_size,
                              hipStream_t stream) {
    const float* x     = (const float*)d_in[0];
    const int*   ei    = (const int*)  d_in[1];
    const float* W1l   = (const float*)d_in[2];
    const float* b1l   = (const float*)d_in[3];
    const float* W1r   = (const float*)d_in[4];
    const float* b1r   = (const float*)d_in[5];
    const float* att1  = (const float*)d_in[6];
    const float* bias1 = (const float*)d_in[7];
    const float* W2l   = (const float*)d_in[8];
    const float* b2l   = (const float*)d_in[9];
    const float* W2r   = (const float*)d_in[10];
    const float* b2r   = (const float*)d_in[11];
    const float* att2  = (const float*)d_in[12];
    const float* bias2 = (const float*)d_in[13];
    const float* Wskip = (const float*)d_in[14];
    const float* bskip = (const float*)d_in[15];
    const float* gamma = (const float*)d_in[16];
    const float* beta  = (const float*)d_in[17];
    float* out = (float*)d_out;

    char* ws = (char*)d_ws;
    size_t off = 0;
    float* XLRS = (float*)(ws + off); off += (size_t)N_NODES * NCOLS * 4;   // 153.6 MB
    bf16*  xbf  = (bf16*) (ws + off); off += (size_t)N_NODES * DIN * 2;     //  25.6 MB
    bf16*  wp   = (bf16*) (ws + off); off += (size_t)KSTEPS * NTILES * 32 * 16 * 2;
    float* lg1  = (float*)(ws + off); off += (size_t)ET * 4 * 4;            //  13.6 MB
    int*   m1   = (int*)  (ws + off); off += (size_t)N_NODES * 4 * 4;
    float* den1 = (float*)(ws + off); off += (size_t)N_NODES * 4 * 4;
    float* h1   = (float*)(ws + off); off += (size_t)N_NODES * HCDIM * 4;   //  51.2 MB
    float* xl2  = (float*)(ws + off); off += (size_t)N_NODES * 2 * 4;
    float* xr2  = (float*)(ws + off); off += (size_t)N_NODES * 2 * 4;
    float* lg2  = (float*)(ws + off); off += (size_t)ET * 4;
    int*   m2   = (int*)  (ws + off); off += (size_t)N_NODES * 4;
    float* den2 = (float*)(ws + off); off += (size_t)N_NODES * 4;

    const int waveBlocksET = (ET + 7) / 8;           // wave-per-edge kernels
    const int gemmWaves = (N_NODES / 16) * NGROUPS;  // 18750

    k_init<<<(N_NODES * HCDIM) / 256, 256, 0, stream>>>(h1, m1, den1, m2, den2, out, bias2);
    k_cvt_x<<<(N_NODES * DIN + 255) / 256, 256, 0, stream>>>(x, xbf);
    k_pack_w<<<(KSTEPS * NTILES * 32 * 16 + 255) / 256, 256, 0, stream>>>(W1l, W1r, Wskip, wp);
    k_gemm1<<<(gemmWaves + 7) / 8, 256, 0, stream>>>(xbf, wp, b1l, b1r, bskip, XLRS);
    k_edge1_logits<<<waveBlocksET, 256, 0, stream>>>(ei, XLRS, att1, lg1, m1);
    k_edge1_denom<<<(ET * 4 + 255) / 256, 256, 0, stream>>>(ei, lg1, m1, den1);
    k_edge1_agg<<<waveBlocksET, 256, 0, stream>>>(ei, XLRS, lg1, den1, h1);
    k_node_fuse<<<N_NODES, 256, 0, stream>>>(h1, XLRS, bias1, gamma, beta);
    k_conv2_lin<<<(N_NODES + 7) / 8, 256, 0, stream>>>(h1, W2l, b2l, W2r, b2r, xl2, xr2);
    k_edge2_logits<<<(ET + 255) / 256, 256, 0, stream>>>(ei, xl2, xr2, att2, lg2, m2);
    k_edge2_denom<<<(ET + 255) / 256, 256, 0, stream>>>(ei, lg2, m2, den2);
    k_edge2_agg<<<(ET + 255) / 256, 256, 0, stream>>>(ei, xl2, lg2, den2, out);
}
